// CAM_28724741276388
// MI455X (gfx1250) — matrix operand order repaired and checked by reading, not compiled, not run on hardware
//
#include <hip/hip_runtime.h>
#include <hip/hip_bf16.h>

// ---- problem constants (from reference) ----
#define BB   16
#define CC   256
#define NHW  4096   // H*W = 64*64

// ---- WMMA vector types (gfx1250, wave32) ----
typedef __bf16 bf16_t;
typedef bf16_t v16bf __attribute__((ext_vector_type(16)));
typedef bf16_t v8bf  __attribute__((ext_vector_type(8)));
typedef float  v8f   __attribute__((ext_vector_type(8)));

// fp32 -> bf16 (round-to-nearest-even) as raw u16 (used in the convert and softmax passes)
static __device__ inline unsigned short f2bfu(float f) {
    unsigned u = __builtin_bit_cast(unsigned, f);
    u += 0x7FFFu + ((u >> 16) & 1u);
    return (unsigned short)(u >> 16);
}
static __device__ inline unsigned pack2(unsigned short lo, unsigned short hi) {
    return (unsigned)lo | ((unsigned)hi << 16);
}
// One v_perm_b32: build [elem q of rowB : elem q of rowA] from packed u32 words.
// perm pool = bytes 0-3 from src1, bytes 4-7 from src0.
static __device__ inline unsigned perm_pair(unsigned wA, unsigned wB, int q) {
    const unsigned sel = (q & 1) ? 0x07060302u : 0x05040100u;
    return __builtin_amdgcn_perm(wB, wA, sel);
}

// Fragment builders (LDS row = matrix row, K contiguous, stride 40 u16 = 80 B).
static __device__ inline v16bf load_frag_A(const unsigned short* rowp, int hi) {
    v8bf lo = *(const v8bf*)(rowp + hi * 8);
    v8bf hh = *(const v8bf*)(rowp + 16 + hi * 8);
    v16bf f;
    #pragma unroll
    for (int q = 0; q < 8; ++q) { f[q] = lo[q]; f[8 + q] = hh[q]; }
    return f;
}
static __device__ inline v16bf load_frag_B(const unsigned short* rowp, int hi) {
    v8bf lo = *(const v8bf*)(rowp + hi * 8);
    v8bf hh = *(const v8bf*)(rowp + 16 + hi * 8);
    v16bf f;
    #pragma unroll
    for (int q = 0; q < 8; ++q) { f[q] = lo[q]; f[8 + q] = hh[q]; }
    return f;
}
#define WMMA_BF16(A, B, C) \
    __builtin_amdgcn_wmma_f32_16x16x32_bf16(false, (A), false, (B), (short)0, (C), false, false)

// =====================================================================
// Kernel 0: one-time fp32 -> bf16 conversion of x (paid once, not 8x).
// Grid: 4096 blocks x 256 threads x 16 elements = 16,777,216
// =====================================================================
__global__ void __launch_bounds__(256)
CAM_cvt_kernel(const float* __restrict__ x, unsigned short* __restrict__ xb)
{
    const size_t base = ((size_t)blockIdx.x * 256 + threadIdx.x) * 16;
    float4 f0 = *(const float4*)(x + base);
    float4 f1 = *(const float4*)(x + base + 4);
    float4 f2 = *(const float4*)(x + base + 8);
    float4 f3 = *(const float4*)(x + base + 12);
    float e[16] = {f0.x,f0.y,f0.z,f0.w, f1.x,f1.y,f1.z,f1.w,
                   f2.x,f2.y,f2.z,f2.w, f3.x,f3.y,f3.z,f3.w};
    unsigned o[8];
    #pragma unroll
    for (int q = 0; q < 8; ++q) o[q] = pack2(f2bfu(e[2 * q]), f2bfu(e[2 * q + 1]));
    *(uint4*)(xb + base)     = make_uint4(o[0], o[1], o[2], o[3]);
    *(uint4*)(xb + base + 8) = make_uint4(o[4], o[5], o[6], o[7]);
}

// =====================================================================
// Kernel 1: aTa[b] = a^T * a  (C x C per batch), reading bf16 x.
// Block tile 64x64, 4 waves each owning 32x32 (4 accums). Double-buffered LDS.
// Grid: (C/64, C/64, B) = (4, 4, 16), Block: 128
// =====================================================================
__global__ void __launch_bounds__(128)
CAM_aTa_kernel(const unsigned short* __restrict__ xb, float* __restrict__ aTa)
{
    __shared__ __align__(16) unsigned short sAT[2][64][40];  // [ci][n]
    __shared__ __align__(16) unsigned short sBT[2][64][40];  // [cj][n]

    const int b   = blockIdx.z;
    const int ci0 = blockIdx.x * 64;
    const int cj0 = blockIdx.y * 64;
    const int t    = threadIdx.x;
    const int wave = t >> 5;
    const int lane = t & 31;
    const int wi = (wave >> 1) * 32;
    const int wj = (wave & 1)  * 32;
    const int hi = lane >> 4;
    const int ml = lane & 15;
    const int p  = t >> 3;               // n rows {2p, 2p+1}
    const int cb = (t & 7) * 8;          // 8 channels

    const unsigned short* xbb = xb + (size_t)b * NHW * CC;
    v8f acc00 = {}, acc01 = {}, acc10 = {}, acc11 = {};

    for (int it = 0; it < NHW / 32; ++it) {
        const int n0  = it * 32;
        const int buf = it & 1;
        const unsigned short* r0 = xbb + (size_t)(n0 + 2 * p) * CC;
        const unsigned short* r1 = r0 + CC;
        const uint4 A0 = *(const uint4*)(r0 + ci0 + cb);   // 8 bf16, row 2p
        const uint4 A1 = *(const uint4*)(r1 + ci0 + cb);   // 8 bf16, row 2p+1
        const uint4 B0 = *(const uint4*)(r0 + cj0 + cb);
        const uint4 B1 = *(const uint4*)(r1 + cj0 + cb);
        unsigned a0w[4] = {A0.x, A0.y, A0.z, A0.w};
        unsigned a1w[4] = {A1.x, A1.y, A1.z, A1.w};
        unsigned b0w[4] = {B0.x, B0.y, B0.z, B0.w};
        unsigned b1w[4] = {B1.x, B1.y, B1.z, B1.w};
        #pragma unroll
        for (int q = 0; q < 8; ++q) {   // one v_perm_b32 + one ds_store_b32 per pair
            *(unsigned*)&sAT[buf][cb + q][2 * p] = perm_pair(a0w[q >> 1], a1w[q >> 1], q);
            *(unsigned*)&sBT[buf][cb + q][2 * p] = perm_pair(b0w[q >> 1], b1w[q >> 1], q);
        }
        if (it + 1 < NHW / 32) {   // global_prefetch_b8 of next K-chunk
            __builtin_prefetch(r0 + (size_t)32 * CC + ci0 + cb, 0, 0);
            __builtin_prefetch(r0 + (size_t)32 * CC + cj0 + cb, 0, 0);
        }
        __syncthreads();

        v16bf a0f = load_frag_A(&sAT[buf][wi + ml][0],      hi);
        v16bf a1f = load_frag_A(&sAT[buf][wi + 16 + ml][0], hi);
        v16bf b0f = load_frag_B(&sBT[buf][wj + ml][0],      hi);
        v16bf b1f = load_frag_B(&sBT[buf][wj + 16 + ml][0], hi);
        acc00 = WMMA_BF16(a0f, b0f, acc00);
        acc01 = WMMA_BF16(a0f, b1f, acc01);
        acc10 = WMMA_BF16(a1f, b0f, acc10);
        acc11 = WMMA_BF16(a1f, b1f, acc11);
    }

    v8f accs[2][2] = {{acc00, acc01}, {acc10, acc11}};
    #pragma unroll
    for (int rt = 0; rt < 2; ++rt)
        #pragma unroll
        for (int ct = 0; ct < 2; ++ct)
            #pragma unroll
            for (int r = 0; r < 8; ++r) {
                size_t o = ((size_t)b * CC + (ci0 + wi + rt * 16 + hi * 8 + r)) * CC
                         + (cj0 + wj + ct * 16 + ml);
                aTa[o] = accs[rt][ct][r];
            }
}

// =====================================================================
// Kernel 2: row softmax of aTa -> attn (bf16). One 256-thread block per row.
// =====================================================================
__global__ void __launch_bounds__(256)
CAM_softmax_kernel(const float* __restrict__ aTa, unsigned short* __restrict__ attn)
{
    __shared__ float red[256];
    const int row = blockIdx.x;
    const int t   = threadIdx.x;
    const float v = aTa[(size_t)row * CC + t];

    red[t] = v; __syncthreads();
    #pragma unroll
    for (int s = 128; s > 0; s >>= 1) {
        if (t < s) red[t] = fmaxf(red[t], red[t + s]);
        __syncthreads();
    }
    const float m = red[0]; __syncthreads();

    const float e = __expf(v - m);
    red[t] = e; __syncthreads();
    #pragma unroll
    for (int s = 128; s > 0; s >>= 1) {
        if (t < s) red[t] += red[t + s];
        __syncthreads();
    }
    attn[(size_t)row * CC + t] = f2bfu(e * (1.0f / red[0]));
}

// =====================================================================
// Kernel 3: out = gamma * (a @ attn) + x.  M = n, K = i, N = j.
// A tile staging is now a pure b128 copy (x already bf16).
// Grid: (NHW/64, C/64, B) = (64, 4, 16), Block: 128
// =====================================================================
__global__ void __launch_bounds__(128)
CAM_av_kernel(const unsigned short* __restrict__ xb, const float* __restrict__ x,
              const unsigned short* __restrict__ attn,
              const float* __restrict__ gamma, float* __restrict__ out)
{
    __shared__ __align__(16) unsigned short sA [2][64][40];  // [n][k] natural
    __shared__ __align__(16) unsigned short sBT[2][64][40];  // [j][k] transposed attn

    const int b   = blockIdx.z;
    const int n0  = blockIdx.x * 64;
    const int cj0 = blockIdx.y * 64;
    const int t    = threadIdx.x;
    const int wave = t >> 5;
    const int lane = t & 31;
    const int wi = (wave >> 1) * 32;
    const int wj = (wave & 1)  * 32;
    const int hi = lane >> 4;
    const int ml = lane & 15;
    const int arow = t >> 1;             // A staging: n row 0..63
    const int kb   = (t & 1) * 16;       // 16 k per thread
    const int p    = t >> 3;             // B staging: i rows {2p, 2p+1}
    const int jb   = (t & 7) * 8;        // 8 j per thread

    const unsigned short* xbb = xb   + (size_t)b * NHW * CC;
    const unsigned short* ab  = attn + (size_t)b * CC * CC;
    v8f acc00 = {}, acc01 = {}, acc10 = {}, acc11 = {};

    for (int it = 0; it < CC / 32; ++it) {
        const int k0  = it * 32;
        const int buf = it & 1;
        // ---- A tile: straight bf16 copy, global b128 -> LDS b128 ----
        const unsigned short* rp = xbb + (size_t)(n0 + arow) * CC + k0 + kb;
        *(uint4*)&sA[buf][arow][kb]     = *(const uint4*)rp;
        *(uint4*)&sA[buf][arow][kb + 8] = *(const uint4*)(rp + 8);

        // ---- B tile: transpose attn [i][j] -> [j][i] via v_perm_b32 pairs ----
        const uint4 u0 = *(const uint4*)(ab + (size_t)(k0 + 2 * p) * CC + cj0 + jb);
        const uint4 u1 = *(const uint4*)(ab + (size_t)(k0 + 2 * p + 1) * CC + cj0 + jb);
        unsigned w0[4] = {u0.x, u0.y, u0.z, u0.w};
        unsigned w1[4] = {u1.x, u1.y, u1.z, u1.w};
        #pragma unroll
        for (int q = 0; q < 8; ++q)
            *(unsigned*)&sBT[buf][jb + q][2 * p] = perm_pair(w0[q >> 1], w1[q >> 1], q);

        if (it + 1 < CC / 32) {
            __builtin_prefetch(rp + 32, 0, 0);
            __builtin_prefetch(ab + (size_t)(k0 + 32 + 2 * p) * CC + cj0 + jb, 0, 0);
        }
        __syncthreads();

        v16bf a0f = load_frag_A(&sA [buf][wi + ml][0],      hi);
        v16bf a1f = load_frag_A(&sA [buf][wi + 16 + ml][0], hi);
        v16bf b0f = load_frag_B(&sBT[buf][wj + ml][0],      hi);
        v16bf b1f = load_frag_B(&sBT[buf][wj + 16 + ml][0], hi);
        acc00 = WMMA_BF16(a0f, b0f, acc00);
        acc01 = WMMA_BF16(a0f, b1f, acc01);
        acc10 = WMMA_BF16(a1f, b0f, acc10);
        acc11 = WMMA_BF16(a1f, b1f, acc11);
    }

    const float g = gamma[0];
    v8f accs[2][2] = {{acc00, acc01}, {acc10, acc11}};
    #pragma unroll
    for (int rt = 0; rt < 2; ++rt)
        #pragma unroll
        for (int ct = 0; ct < 2; ++ct)
            #pragma unroll
            for (int r = 0; r < 8; ++r) {
                size_t o = ((size_t)b * NHW + (n0 + wi + rt * 16 + hi * 8 + r)) * CC
                         + (cj0 + wj + ct * 16 + ml);
                out[o] = g * accs[rt][ct][r] + x[o];   // residual uses original fp32 x
            }
}

// =====================================================================
// Host-side launcher
// =====================================================================
extern "C" void kernel_launch(void* const* d_in, const int* in_sizes, int n_in,
                              void* d_out, int out_size, void* d_ws, size_t ws_size,
                              hipStream_t stream) {
    (void)in_sizes; (void)n_in; (void)out_size; (void)ws_size;
    const float* x     = (const float*)d_in[0];
    const float* gamma = (const float*)d_in[1];
    float*       out   = (float*)d_out;

    // workspace: [ xb : 32 MiB bf16 ][ aTa : 4 MiB f32 ][ attn : 2 MiB bf16 ]
    const size_t NTOT = (size_t)BB * NHW * CC;               // 16,777,216
    unsigned short* xbf  = (unsigned short*)d_ws;
    float*          aTa  = (float*)((char*)d_ws + NTOT * 2);
    unsigned short* attn = (unsigned short*)((char*)d_ws + NTOT * 2
                                             + (size_t)BB * CC * CC * sizeof(float));

    CAM_cvt_kernel<<<(int)(NTOT / (256 * 16)), 256, 0, stream>>>(x, xbf);

    dim3 g1(CC / 64, CC / 64, BB);            // (4, 4, 16)
    CAM_aTa_kernel<<<g1, 128, 0, stream>>>(xbf, aTa);

    CAM_softmax_kernel<<<BB * CC, 256, 0, stream>>>(aTa, attn);

    dim3 g2(NHW / 64, CC / 64, BB);           // (64, 4, 16)
    CAM_av_kernel<<<g2, 128, 0, stream>>>(xbf, x, attn, gamma, out);
}
